// MultiHeadAttentionMVI_72756745994363
// MI455X (gfx1250) — matrix operand order repaired and checked by reading, not compiled, not run on hardware
//
#include <hip/hip_runtime.h>
#include <stdint.h>

typedef __bf16 bf16;
typedef __attribute__((ext_vector_type(16))) __bf16 v16bf;
typedef __attribute__((ext_vector_type(8)))  float  v8f;

// ---------------- problem constants ----------------
#define NWIN   256      // n windows
#define N2     254      // n-2 valid rows
#define WIN    10
#define DM     512
#define NH     8
#define DEPTH  64
#define SLEN   2560
#define THREADS 128
#define WAVES   4

// ---------------- LDS layout (bytes) ----------------
#define OFF_QIN   0                       // bf16 [257][48] qk_in padded (v-in aliased via +10 col offset)
#define SZ_QIN    (257*48*2)              // 24672
#define OFF_MEAN  (OFF_QIN + SZ_QIN)      // f32 [256]
#define OFF_PAD   (OFF_MEAN + 1024)       // f32 [256] padding-mask additive (-1e9 / 0)
#define OFF_MFLAG (OFF_PAD + 1024)        // u32 [256] mask-min flags
#define OFF_KH    (OFF_MFLAG + 1024)      // bf16 [256][64] K (row major)
#define OFF_VT    (OFF_KH + 32768)        // bf16 [64][260] V transposed
#define OFF_WQ    (OFF_VT + 33280)        // bf16 [64][32] Wq^T slice (K padded)
#define OFF_WK    (OFF_WQ + 4096)
#define OFF_WV    (OFF_WK + 4096)
#define OFF_WD    (OFF_WV + 4096)         // f32 [64][10] dense slice
#define OFF_QH    (OFF_WD + 2560)         // bf16 [256][64] Q ; aliases f32 dat[256][10] in setup
#define OFF_OST   (OFF_QH + 32768)        // per-wave bf16 [16][64] O staging; P staging aliased inside
#define OFF_OACC  (OFF_OST + WAVES*2048)  // f32 [256][10] output accumulator
#define SMEM_SZ   (OFF_OACC + 10240)      // 159840 bytes (~156 KB)

union FragU { uint32_t u[8]; v16bf v; };

static __device__ __forceinline__ v8f wmma_bf16(v16bf a, v16bf b, v8f c) {
    return __builtin_amdgcn_wmma_f32_16x16x32_bf16(false, a, false, b, (short)0, c, false, false);
}

// A-matrix (16x32 bf16, M x K) fragment. base -> element [0][0], strideElems = row stride.
// Lane m=lane&15 holds row M=m; g=lane>>4 selects K half per ISA layout.
static __device__ __forceinline__ v16bf load_a(const bf16* base, int strideElems) {
    const int lane = threadIdx.x & 31;
    const int g = lane >> 4, m = lane & 15;
    const bf16* p = base + m * strideElems;
    FragU f;
#pragma unroll
    for (int r = 0; r < 8; ++r) {
        int K = (r < 4) ? (8 * g + 2 * r) : (16 + 8 * g + 2 * (r - 4));
        f.u[r] = *(const uint32_t*)(p + K);
    }
    return f.v;
}

// B-matrix (32x16 bf16, K x N) fragment from memory laid out [N][K] (K contiguous).
static __device__ __forceinline__ v16bf load_b(const bf16* base, int strideElems) {
    const int lane = threadIdx.x & 31;
    const int g = lane >> 4, n = lane & 15;
    const bf16* p = base + n * strideElems;
    FragU f;
#pragma unroll
    for (int r = 0; r < 8; ++r) {
        int K = (r < 4) ? (8 * g + 2 * r) : (16 + 8 * g + 2 * (r - 4));
        f.u[r] = *(const uint32_t*)(p + K);
    }
    return f.v;
}

__global__ __launch_bounds__(THREADS, 1)
void mha_mvi_kernel(const float* __restrict__ x,
                    const float* __restrict__ wq, const float* __restrict__ wqb,
                    const float* __restrict__ wk, const float* __restrict__ wkb,
                    const float* __restrict__ wvw, const float* __restrict__ wvb,
                    const float* __restrict__ dw, const float* __restrict__ db,
                    float* __restrict__ out) {
    __shared__ __align__(16) unsigned char smem[SMEM_SZ];
    bf16*     qin   = (bf16*)(smem + OFF_QIN);
    float*    meanS = (float*)(smem + OFF_MEAN);
    float*    padS  = (float*)(smem + OFF_PAD);
    unsigned* mflag = (unsigned*)(smem + OFF_MFLAG);
    bf16*     Kh    = (bf16*)(smem + OFF_KH);
    bf16*     Vt    = (bf16*)(smem + OFF_VT);
    bf16*     Wq    = (bf16*)(smem + OFF_WQ);
    bf16*     Wk    = (bf16*)(smem + OFF_WK);
    bf16*     Wv    = (bf16*)(smem + OFF_WV);
    float*    Wd    = (float*)(smem + OFF_WD);
    bf16*     Qh    = (bf16*)(smem + OFF_QH);
    float*    dat   = (float*)(smem + OFF_QH);   // alias: dead before Qh is written
    bf16*     OstB  = (bf16*)(smem + OFF_OST);
    float*    oacc  = (float*)(smem + OFF_OACC);

    const int tid  = threadIdx.x;
    const int lane = tid & 31;
    // wave id as a scalar -> all job/tile loops below run on SGPR induction
    // variables and all t/m/nt branches become scalar branches (no saveexec).
    const int wids = __builtin_amdgcn_readfirstlane(tid >> 5);
    const int n16  = lane & 15;
    const int g16  = lane >> 4;

    // warm GL2 for this block's x slab (emits global_prefetch_b8)
    const size_t xb = (size_t)blockIdx.x * (SLEN * 3);
    {
        const char* xc = (const char*)(x + xb);
        for (int o = tid * 128; o < SLEN * 3 * 4; o += THREADS * 128)
            __builtin_prefetch(xc + o, 0, 1);
    }

    // ---- phase 0: init accumulators ----
    for (int i = tid; i < NWIN; i += THREADS) { meanS[i] = 0.f; mflag[i] = 0xFFFFFFFFu; }
    for (int i = tid; i < NWIN * WIN; i += THREADS) oacc[i] = 0.f;
    __syncthreads();

    // ---- phase 1: load x, masked means, mask flags, data+PE ----
    for (int s = tid; s < SLEN; s += THREADS) {
        float d  = x[xb + 3 * s];
        float mk = x[xb + 3 * s + 2];
        int i = s / WIN, w = s % WIN;
        atomicAdd(&meanS[i], d * mk * 0.1f);
        if (mk == 0.0f) atomicAnd(&mflag[i], 0u);
        float e   = (float)((w >> 1) * 2) * 0.1f;         // 2*floor(w/2)/d_model
        float ang = (float)i * __powf(10000.0f, -e);
        float pe  = (w & 1) ? cosf(ang) : sinf(ang);
        dat[i * WIN + w] = d + pe;
    }
    __syncthreads();

    // ---- phase 2: padding-mask additive + qk_in (bf16, K padded to 48 cols) ----
    for (int kk = tid; kk < NWIN; kk += THREADS)
        padS[kk] = (kk <= 253 && mflag[kk + 1] != 0u) ? 0.0f : -1e9f;
    for (int idx = tid; idx < 257 * 48; idx += THREADS) {
        int row = idx / 48, col = idx % 48;
        float v = 0.0f;
        if (row < 256) {
            if (col < 10)      { if (row <= 253) v = dat[(row + 2) * WIN + col]; }
            else if (col < 20) { v = dat[row * WIN + (col - 10)]; }
        }
        qin[idx] = (bf16)v;
    }
    __syncthreads();   // dat dead from here; Qh region reusable

    const v8f Z = {0.f,0.f,0.f,0.f,0.f,0.f,0.f,0.f};

    // ================= per-head loop =================
    for (int h = 0; h < NH; ++h) {
        __syncthreads();   // protect Kh/Vt/Qh/W reuse across heads
        // weight slices: W[d][k] transposed, K-dim zero-padded to 32
        for (int idx = tid; idx < 64 * 32; idx += THREADS) {
            int d = idx >> 5, k = idx & 31;
            Wq[idx] = (bf16)((k < 20) ? wq [k * DM + h * DEPTH + d] : 0.f);
            Wk[idx] = (bf16)((k < 20) ? wk [k * DM + h * DEPTH + d] : 0.f);
            Wv[idx] = (bf16)((k < 10) ? wvw[k * DM + h * DEPTH + d] : 0.f);
        }
        for (int idx = tid; idx < 64 * WIN; idx += THREADS) {
            int d = idx / WIN, w = idx % WIN;
            Wd[idx] = dw[(h * DEPTH + d) * WIN + w];
        }
        __syncthreads();

        // ---- projections: Q,K,V via WMMA (16 Mtiles x 4 Ntiles x {Q,K,V}) ----
        for (int job = wids; job < 192; job += WAVES) {
            const int t = job >> 6, rem = job & 63, m = rem >> 2, nt = rem & 3;
            const bf16* abase = (t < 2) ? (qin + m * 16 * 48)
                                        : (qin + (m * 16 + 1) * 48 + 10);
            v16bf a = load_a(abase, 48);
            const bf16* wsel = (t == 0) ? Wq : (t == 1) ? Wk : Wv;
            v16bf b = load_b(wsel + nt * 16 * 32, 32);
            const float* bias = (t == 0) ? wqb : (t == 1) ? wkb : wvb;
            float bv = bias[h * DEPTH + nt * 16 + n16];
            v8f c = {bv,bv,bv,bv,bv,bv,bv,bv};
            c = wmma_bf16(a, b, c);
            // t is scalar -> uniform branch, hoisted around the whole store loop
            if (t == 0) {
#pragma unroll
                for (int r = 0; r < 8; ++r)
                    Qh[(m * 16 + r + 8 * g16) * 64 + nt * 16 + n16] = (bf16)c[r];
            } else if (t == 1) {
#pragma unroll
                for (int r = 0; r < 8; ++r)
                    Kh[(m * 16 + r + 8 * g16) * 64 + nt * 16 + n16] = (bf16)c[r];
            } else {
#pragma unroll
                for (int r = 0; r < 8; ++r)
                    Vt[(nt * 16 + n16) * 260 + m * 16 + r + 8 * g16] = (bf16)c[r];
            }
        }
        __syncthreads();

        // ---- flash attention: each wave owns q-tiles wids, wids+4, ... ----
        bf16* Pst = OstB + wids * 1024;   // [16][32] staging (aliased inside Ost region)
        bf16* Ost = OstB + wids * 1024;   // [16][64] bf16 O staging
        for (int qt = wids; qt < 16; qt += WAVES) {
            v16bf aq0 = load_a(Qh + qt * 16 * 64,      64);
            v16bf aq1 = load_a(Qh + qt * 16 * 64 + 32, 64);
            float mrow[8], lrow[8];
            v8f o0 = Z, o1 = Z, o2 = Z, o3 = Z;
#pragma unroll
            for (int r = 0; r < 8; ++r) { mrow[r] = -3.0e38f; lrow[r] = 0.f; }

            const int kbMax = qt >> 1;   // causal: keys kk <= 16*qt+15
            for (int kb = 0; kb <= kbMax; ++kb) {
                int kk0 = kb * 32;
                v8f s0 = Z, s1 = Z;
                s0 = wmma_bf16(aq0, load_b(Kh + kk0 * 64,             64), s0);
                s0 = wmma_bf16(aq1, load_b(Kh + kk0 * 64 + 32,        64), s0);
                s1 = wmma_bf16(aq0, load_b(Kh + (kk0 + 16) * 64,      64), s1);
                s1 = wmma_bf16(aq1, load_b(Kh + (kk0 + 16) * 64 + 32, 64), s1);
                float pad0 = padS[kk0 + n16];
                float pad1 = padS[kk0 + 16 + n16];
#pragma unroll
                for (int r = 0; r < 8; ++r) {
                    int q = qt * 16 + r + 8 * g16;
                    float v0 = s0[r] * 0.125f + ((kk0 + n16      > q) ? -1e9f : pad0);
                    float v1 = s1[r] * 0.125f + ((kk0 + 16 + n16 > q) ? -1e9f : pad1);
                    float bm = fmaxf(v0, v1);
                    bm = fmaxf(bm, __shfl_xor(bm, 1));
                    bm = fmaxf(bm, __shfl_xor(bm, 2));
                    bm = fmaxf(bm, __shfl_xor(bm, 4));
                    bm = fmaxf(bm, __shfl_xor(bm, 8));
                    float mnew = fmaxf(mrow[r], bm);
                    float sc = __expf(mrow[r] - mnew);
                    float p0 = __expf(v0 - mnew);
                    float p1 = __expf(v1 - mnew);
                    float ps = p0 + p1;
                    ps += __shfl_xor(ps, 1);
                    ps += __shfl_xor(ps, 2);
                    ps += __shfl_xor(ps, 4);
                    ps += __shfl_xor(ps, 8);
                    lrow[r] = lrow[r] * sc + ps;
                    mrow[r] = mnew;
                    o0[r] *= sc; o1[r] *= sc; o2[r] *= sc; o3[r] *= sc;
                    Pst[(r + 8 * g16) * 32 + n16]      = (bf16)p0;
                    Pst[(r + 8 * g16) * 32 + 16 + n16] = (bf16)p1;
                }
                v16bf ap = load_a(Pst, 32);
                o0 = wmma_bf16(ap, load_b(Vt + 0 * 16 * 260 + kk0, 260), o0);
                o1 = wmma_bf16(ap, load_b(Vt + 1 * 16 * 260 + kk0, 260), o1);
                o2 = wmma_bf16(ap, load_b(Vt + 2 * 16 * 260 + kk0, 260), o2);
                o3 = wmma_bf16(ap, load_b(Vt + 3 * 16 * 260 + kk0, 260), o3);
            }
            // normalize + stage O
#pragma unroll
            for (int r = 0; r < 8; ++r) {
                float inv = 1.0f / lrow[r];
                int row = r + 8 * g16;
                Ost[row * 64 +      n16] = (bf16)(o0[r] * inv);
                Ost[row * 64 + 16 + n16] = (bf16)(o1[r] * inv);
                Ost[row * 64 + 32 + n16] = (bf16)(o2[r] * inv);
                Ost[row * 64 + 48 + n16] = (bf16)(o3[r] * inv);
            }
            // fold O[16,64] through dense slice into the output accumulator
#pragma unroll
            for (int e = 0; e < 5; ++e) {
                int idx = lane * 5 + e;
                int row = idx / WIN, w = idx % WIN;
                float acc = 0.f;
#pragma unroll 8
                for (int d = 0; d < DEPTH; ++d)
                    acc += (float)Ost[row * 64 + d] * Wd[d * WIN + w];
                oacc[(qt * 16 + row) * WIN + w] += acc;
            }
        }
    }
    __syncthreads();

    // ---- final: add mean-feature term + bias, write out ----
    const size_t ob = (size_t)blockIdx.x * (N2 * WIN);
    for (int idx = tid; idx < N2 * WIN; idx += THREADS) {
        int i = idx / WIN, w = idx % WIN;
        out[ob + idx] = oacc[idx] + meanS[i + 1] * dw[DM * WIN + w] + db[w];
    }
}

extern "C" void kernel_launch(void* const* d_in, const int* in_sizes, int n_in,
                              void* d_out, int out_size, void* d_ws, size_t ws_size,
                              hipStream_t stream) {
    (void)in_sizes; (void)n_in; (void)out_size; (void)d_ws; (void)ws_size;
    const float* x   = (const float*)d_in[0];
    const float* wq  = (const float*)d_in[1];
    const float* wqb = (const float*)d_in[2];
    const float* wk  = (const float*)d_in[3];
    const float* wkb = (const float*)d_in[4];
    const float* wvw = (const float*)d_in[5];
    const float* wvb = (const float*)d_in[6];
    const float* dw  = (const float*)d_in[7];
    const float* db  = (const float*)d_in[8];
    float* out = (float*)d_out;
    mha_mvi_kernel<<<dim3(256), dim3(THREADS), 0, stream>>>(
        x, wq, wqb, wk, wkb, wvw, wvb, dw, db, out);
}
